// CATKT_37340445671985
// MI455X (gfx1250) — matrix operand order repaired and checked by reading, not compiled, not run on hardware
//
#include <hip/hip_runtime.h>
#include <cstdint>
#include <cstddef>

typedef __bf16 bf16;
typedef __attribute__((ext_vector_type(16))) __bf16 v16bf;
typedef __attribute__((ext_vector_type(8)))  __bf16 v8bf;
typedef __attribute__((ext_vector_type(8)))  float  v8f;
typedef __attribute__((ext_vector_type(4)))  int    v4i;

#define B_   32
#define L_   1024
#define NUMC 1000
#define HID  512
#define G4   2048   // 4*HID
#define ATTD 80

// ---- gfx1250 async global->LDS staging (builtin confirmed: v4i AS1 -> v4i AS3) ----
#if defined(__AMDGCN__) && __has_builtin(__builtin_amdgcn_global_load_async_to_lds_b128)
  #define HAVE_ASYNC 1
  static __device__ __forceinline__ void async_cp16(const bf16* g, bf16* l){
    __builtin_amdgcn_global_load_async_to_lds_b128(
        (__attribute__((address_space(1))) v4i*)g,
        (__attribute__((address_space(3))) v4i*)l, 0, 0);
  }
  static __device__ __forceinline__ void wait_async0(){
  #if __has_builtin(__builtin_amdgcn_s_wait_asynccnt)
    __builtin_amdgcn_s_wait_asynccnt(0);
  #else
    asm volatile("s_wait_asynccnt 0" ::: "memory");
  #endif
  }
#else
  static __device__ __forceinline__ void wait_async0(){}
#endif

static __device__ __forceinline__ float sigmoidf_(float x){ return 1.0f/(1.0f+__expf(-x)); }

// A fragment (16x32 bf16), ISA layout: lanes<16 hold K 0-7 & 16-23; lanes>=16 hold K 8-15 & 24-31.
static __device__ __forceinline__ v16bf frag_a(const bf16* p){
  v8bf lo = *(const v8bf*)(p);
  v8bf hi = *(const v8bf*)(p + 16);
  v16bf r;
  #pragma unroll
  for (int i = 0; i < 8; ++i){ r[i] = lo[i]; r[i+8] = hi[i]; }
  return r;
}
// B fragment (32x16 bf16): lanes<16 hold K 0-7 & 16-23 of column n=lane, lanes>=16 hold K 8-15 & 24-31.
static __device__ __forceinline__ v16bf frag_b(const bf16* p){
  v8bf lo = *(const v8bf*)(p);
  v8bf hi = *(const v8bf*)(p + 16);
  v16bf r;
  #pragma unroll
  for (int i = 0; i < 8; ++i){ r[i] = lo[i]; r[i+8] = hi[i]; }
  return r;
}

// ---------------- setup kernels ----------------

__global__ void init_kernel(const float* __restrict__ b_ih, const float* __restrict__ b_hh,
                            const float* __restrict__ fc_b, float* __restrict__ bsum,
                            float* __restrict__ fcbp, bf16* __restrict__ hbuf,
                            unsigned* __restrict__ cnt){
  int t = threadIdx.x;
  for (int i = t; i < G4; i += 256)      bsum[i] = b_ih[i] + b_hh[i];
  for (int i = t; i < 1024; i += 256)    fcbp[i] = (i < NUMC) ? fc_b[i] : 0.0f;
  for (int i = t; i < 32*HID; i += 256)  hbuf[i] = (bf16)0.0f;   // h phase-0 = 0
  if (t < 2) cnt[t] = 0u;
}

__global__ void f32_to_bf16_k(const float* __restrict__ s, bf16* __restrict__ d, int n){
  int i = (blockIdx.x * 256 + threadIdx.x) * 4;
  if (i + 3 < n){
    #pragma unroll
    for (int j = 0; j < 4; ++j) d[i+j] = (bf16)s[i+j];
  }
}

__global__ void fcw_pad_k(const float* __restrict__ fcW, bf16* __restrict__ dst){
  int r = blockIdx.x;           // 0..1023 (padded N)
  int t = threadIdx.x;
  for (int c = t; c < 1024; c += 256)
    dst[(size_t)r*1024 + c] = (r < NUMC) ? (bf16)fcW[(size_t)r*1024 + c] : (bf16)0.0f;
}

__global__ void embed_kernel(const int* __restrict__ skill, const int* __restrict__ answer,
                             const float* __restrict__ st, const float* __restrict__ at,
                             bf16* __restrict__ sae){
  int row = blockIdx.x;         // b*L + l
  int t   = threadIdx.x;        // 0..255
  int s = skill[row];
  int a = answer[row];
  const float* se = st + (size_t)s * 256;
  const float* ae = at + (size_t)a * 256;
  const float* f0 = (a == 1) ? se : ae;
  const float* f1 = (a == 1) ? ae : se;
  bf16* o = sae + (size_t)row * 512;
  o[t]       = (bf16)f0[t];
  o[256 + t] = (bf16)f1[t];
}

// ---------------- WMMA GEMM: C = act(A(MxK) @ Bw(NxK)^T + bias) ----------------
// 128x128 tile, BK=32, 8 waves: 64x32 per wave (4x2 16x16 WMMA tiles).
// Double-buffered LDS staging (async global->LDS).

template<bool OUT_BF16, bool SIGMOID>
__global__ void gemm_wmma(const bf16* __restrict__ A, const bf16* __restrict__ Bw,
                          const float* __restrict__ bias, void* __restrict__ Cp,
                          int M, int N, int K, int ldc, int nmax){
  (void)M; (void)N;
  extern __shared__ __align__(16) char smem[];
  bf16* Asm = (bf16*)smem;                 // [2][128*40] halfs = 20480 B
  bf16* Bsm = (bf16*)(smem + 20480);       // [2][128*40] halfs = 20480 B
  const int t    = threadIdx.x;
  const int m0   = blockIdx.y * 128;
  const int n0   = blockIdx.x * 128;
  const int w    = t >> 5, lane = t & 31;
  const int m_off = (w >> 2) * 64;
  const int n_off = (w & 3) * 32;
  const int r  = lane & 15, hiL = lane >> 4;
  const int ac0 = (lane < 16) ? 0 : 8;
  const int bc0 = (lane < 16) ? 0 : 8;
  const int lrow = t >> 1;
  const int lkc  = (t & 1) * 16;

  auto stage = [&](int buf, int k0){
    const bf16* ga = A  + (size_t)(m0 + lrow) * K + k0 + lkc;
    const bf16* gb = Bw + (size_t)(n0 + lrow) * K + k0 + lkc;
    bf16* la = Asm + buf*5120 + lrow*40 + lkc;
    bf16* lb = Bsm + buf*5120 + lrow*40 + lkc;
#if defined(HAVE_ASYNC)
    async_cp16(ga,     la);
    async_cp16(ga + 8, la + 8);
    async_cp16(gb,     lb);
    async_cp16(gb + 8, lb + 8);
#else
    v8bf a0 = *(const v8bf*)ga;  v8bf a1 = *(const v8bf*)(ga + 8);
    v8bf b0 = *(const v8bf*)gb;  v8bf b1 = *(const v8bf*)(gb + 8);
    *(v8bf*)la       = a0;  *(v8bf*)(la + 8) = a1;
    *(v8bf*)lb       = b0;  *(v8bf*)(lb + 8) = b1;
    if (k0 + 32 < K){ __builtin_prefetch(ga + 32, 0, 1); __builtin_prefetch(gb + 32, 0, 1); }
#endif
  };

  const v8f vzero = {0.f,0.f,0.f,0.f,0.f,0.f,0.f,0.f};
  v8f acc[4][2];
  #pragma unroll
  for (int mi = 0; mi < 4; ++mi)
    #pragma unroll
    for (int ni = 0; ni < 2; ++ni) acc[mi][ni] = vzero;

  stage(0, 0);
  wait_async0();
  __syncthreads();

  int buf = 0;
  for (int k0 = 0; k0 < K; k0 += 32, buf ^= 1){
    if (k0 + 32 < K) stage(buf ^ 1, k0 + 32);

    const bf16* Ab = Asm + buf*5120;
    const bf16* Bb = Bsm + buf*5120;
    v16bf af[4], bfm[2];
    #pragma unroll
    for (int mi = 0; mi < 4; ++mi) af[mi]  = frag_a(&Ab[(m_off + mi*16 + r)*40 + ac0]);
    #pragma unroll
    for (int ni = 0; ni < 2; ++ni) bfm[ni] = frag_b(&Bb[(n_off + ni*16 + r)*40 + bc0]);
    #pragma unroll
    for (int mi = 0; mi < 4; ++mi)
      #pragma unroll
      for (int ni = 0; ni < 2; ++ni)
        acc[mi][ni] = __builtin_amdgcn_wmma_f32_16x16x32_bf16(
            false, af[mi], false, bfm[ni], (short)0, acc[mi][ni], false, false);

    wait_async0();
    __syncthreads();
  }

  #pragma unroll
  for (int mi = 0; mi < 4; ++mi){
    #pragma unroll
    for (int ni = 0; ni < 2; ++ni){
      int col = n0 + n_off + ni*16 + r;
      float bc = bias[col];
      #pragma unroll
      for (int vr = 0; vr < 8; ++vr){
        int m = m0 + m_off + mi*16 + vr + 8*hiL;
        float v = acc[mi][ni][vr] + bc;
        if (SIGMOID) v = 1.0f / (1.0f + __expf(-v));
        if (OUT_BF16){
          ((bf16*)Cp)[(size_t)m * ldc + col] = (bf16)v;
        } else {
          if (col < nmax) ((float*)Cp)[(size_t)m * ldc + col] = v;
        }
      }
    }
  }
}

// ---------------- persistent WMMA LSTM ----------------
// 32 WGs: group = batch half (16 rows), part = 32-wide hidden-column slice.
// W_hh slice (128x512 bf16) resident in LDS; h double-buffered in global,
// steps synchronized with a monotonic agent-scope atomic barrier.
// Two independent WMMA accumulators (even/odd K) halve the dependency chain.

__global__ void lstm_kernel(const bf16* __restrict__ whh, const bf16* __restrict__ xpre,
                            bf16* __restrict__ hbuf, float* __restrict__ outf,
                            unsigned* __restrict__ cnt){
  extern __shared__ __align__(16) char smem[];
  bf16*  whh_s = (bf16*)smem;                                   // 128 rows, stride 520
  bf16*  h_s   = (bf16*)(smem + 133120);                        // 16 rows, stride 520
  float* gbuf  = (float*)(smem + 133120 + 16640);               // 4*16*32 f32
  float* c_s   = (float*)(smem + 133120 + 16640 + 8192);        // 16*32 f32

  const int t     = threadIdx.x;
  const int group = blockIdx.x >> 4;
  const int part  = blockIdx.x & 15;
  const int w = t >> 5, lane = t & 31;
  const int r = lane & 15, hiL = lane >> 4;
  const int ac0 = (lane < 16) ? 0 : 8;
  const int bc0 = (lane < 16) ? 0 : 8;
  const int g = w >> 1, nh = w & 1;

  // stage the W_hh slice for this (part): rows g'*512 + part*32 + c, c=0..31, g'=0..3
  for (int i = 0; i < 32; ++i){
    int idx = t + i*256;                 // 0..8191 chunks of 8 halfs
    int s  = idx >> 6;                   // slice row 0..127
    int kc = (idx & 63) * 8;             // 0..504
    int R  = (s >> 5)*HID + part*32 + (s & 31);
#if defined(HAVE_ASYNC)
    async_cp16(whh + (size_t)R*HID + kc, &whh_s[s*520 + kc]);
#else
    v8bf v = *(const v8bf*)(whh + (size_t)R*HID + kc);
    *(v8bf*)&whh_s[s*520 + kc] = v;
#endif
  }
  for (int i = t; i < 512; i += 256) c_s[i] = 0.0f;
  wait_async0();
  __syncthreads();

  const size_t hstride = (size_t)32 * 512;          // per phase
  bf16* hb = hbuf + (size_t)group * 16 * 512;
  const int col = g*512 + part*32 + nh*16 + r;      // global gate column for this lane
  const v8f vzero = {0.f,0.f,0.f,0.f,0.f,0.f,0.f,0.f};

  for (int l = 0; l < L_; ++l){
    const bf16* hprev = hb + (size_t)(l & 1) * hstride;
    // stage h (16x512 bf16) into LDS
    #pragma unroll
    for (int i = 0; i < 4; ++i){
      int idx = t + i*256;
      int hr = idx >> 6;  int kc = (idx & 63) * 8;
#if defined(HAVE_ASYNC)
      async_cp16(hprev + (size_t)hr*512 + kc, &h_s[hr*520 + kc]);
#else
      v8bf v = *(const v8bf*)(hprev + (size_t)hr*512 + kc);
      *(v8bf*)&h_s[hr*520 + kc] = v;
#endif
    }
    wait_async0();
    __syncthreads();

    v8f acc0 = vzero, acc1 = vzero;
    #pragma unroll
    for (int kk = 0; kk < 16; kk += 2){
      int kb0 = kk * 32;
      int kb1 = kb0 + 32;
      v16bf af0  = frag_a(&h_s[r*520 + kb0 + ac0]);
      v16bf bfm0 = frag_b(&whh_s[(w*16 + r)*520 + kb0 + bc0]);
      acc0 = __builtin_amdgcn_wmma_f32_16x16x32_bf16(false, af0, false, bfm0,
                                                     (short)0, acc0, false, false);
      v16bf af1  = frag_a(&h_s[r*520 + kb1 + ac0]);
      v16bf bfm1 = frag_b(&whh_s[(w*16 + r)*520 + kb1 + bc0]);
      acc1 = __builtin_amdgcn_wmma_f32_16x16x32_bf16(false, af1, false, bfm1,
                                                     (short)0, acc1, false, false);
    }
    v8f acc = acc0 + acc1;

    // add x_pre and stash pre-activations in LDS
    #pragma unroll
    for (int vr = 0; vr < 8; ++vr){
      int m = vr + 8*hiL;
      int b = group*16 + m;
      float xp = (float)xpre[((size_t)b*L_ + l)*G4 + col];
      gbuf[(g*16 + m)*32 + nh*16 + r] = acc[vr] + xp;
    }
    __syncthreads();

    // gate activations + state update (this WG owns h columns part*32..+31)
    bf16* hnext = hb + (size_t)((l + 1) & 1) * hstride;
    #pragma unroll
    for (int e = 0; e < 2; ++e){
      int idx = t + e*256;
      int m = idx >> 5, c = idx & 31;
      float gi = sigmoidf_(gbuf[( 0 + m)*32 + c]);
      float gf = sigmoidf_(gbuf[(16 + m)*32 + c]);
      float gg = tanhf    (gbuf[(32 + m)*32 + c]);
      float go = sigmoidf_(gbuf[(48 + m)*32 + c]);
      float ccv = gf * c_s[m*32 + c] + gi * gg;
      c_s[m*32 + c] = ccv;
      float hh = go * tanhf(ccv);
      int b  = group*16 + m;
      int hc = part*32 + c;
      outf[((size_t)b*L_ + l)*HID + hc] = hh;
      hnext[(size_t)m*512 + hc] = (bf16)hh;
    }
    __threadfence();
    __syncthreads();
    if (t == 0){
      __hip_atomic_fetch_add(&cnt[group], 1u, __ATOMIC_RELEASE, __HIP_MEMORY_SCOPE_AGENT);
      unsigned target = 16u * (unsigned)(l + 1);
      while (__hip_atomic_load(&cnt[group], __ATOMIC_ACQUIRE, __HIP_MEMORY_SCOPE_AGENT) < target)
        __builtin_amdgcn_s_sleep(2);
    }
    __syncthreads();
  }
}

// ---------------- attention score: att = tanh(out @ mlpW^T + b) @ simW^T ----------------

__global__ void att_kernel(const float* __restrict__ outf, const float* __restrict__ mlpW,
                           const float* __restrict__ mlpb, const float* __restrict__ simW,
                           float* __restrict__ att){
  int w = threadIdx.x >> 5, lane = threadIdx.x & 31;
  int row = blockIdx.x * 8 + w;
  const float* x  = outf + (size_t)row * HID;
  int j0 = lane, j1 = lane + 32, j2 = lane + 64;    // j2 valid only for lane<16
  const float* w0 = mlpW + (size_t)j0 * HID;
  const float* w1 = mlpW + (size_t)j1 * HID;
  const float* w2 = mlpW + (size_t)j2 * HID;
  float a0 = 0.f, a1 = 0.f, a2 = 0.f;
  for (int k = 0; k < HID; ++k){
    float xv = x[k];
    a0 += xv * w0[k];
    a1 += xv * w1[k];
    if (lane < 16) a2 += xv * w2[k];
  }
  float v = tanhf(a0 + mlpb[j0]) * simW[j0] + tanhf(a1 + mlpb[j1]) * simW[j1];
  if (lane < 16) v += tanhf(a2 + mlpb[j2]) * simW[j2];
  #pragma unroll
  for (int off = 16; off > 0; off >>= 1) v += __shfl_down(v, off, 32);
  if (lane == 0) att[row] = v;
}

// ---------------- causal-softmax-as-scan + final assembly (bf16) ----------------
// attn_out[l] = cumsum(e*out)/cumsum(e); attn_cum_1[l] = exclusive prefix of attn_out.

__global__ void scan_kernel(const float* __restrict__ outf, const float* __restrict__ att,
                            bf16* __restrict__ fin){
  int b = blockIdx.x, t = threadIdx.x;
  float num0 = 0.f, num1 = 0.f, cum0 = 0.f, cum1 = 0.f, den = 0.f;
  for (int l = 0; l < L_; ++l){
    float e = __expf(att[b*L_ + l]);
    den += e;
    size_t ob = ((size_t)b*L_ + l) * HID;
    float o0 = outf[ob + t], o1 = outf[ob + t + 256];
    size_t fb = ((size_t)b*L_ + l) * 1024;
    fin[fb + t]             = (bf16)cum0;
    fin[fb + t + 256]       = (bf16)cum1;
    fin[fb + 512 + t]       = (bf16)o0;
    fin[fb + 512 + t + 256] = (bf16)o1;
    num0 += e * o0;  num1 += e * o1;
    float inv = 1.0f / den;
    cum0 += num0 * inv;  cum1 += num1 * inv;
  }
}

// ---------------- host ----------------

extern "C" void kernel_launch(void* const* d_in, const int* in_sizes, int n_in,
                              void* d_out, int out_size, void* d_ws, size_t ws_size,
                              hipStream_t stream){
  (void)in_sizes; (void)n_in; (void)out_size; (void)ws_size;
  const int*   skill  = (const int*)  d_in[0];
  const int*   answer = (const int*)  d_in[1];
  const float* st     = (const float*)d_in[2];
  const float* at     = (const float*)d_in[3];
  const float* W_ih   = (const float*)d_in[4];
  const float* W_hh   = (const float*)d_in[5];
  const float* b_ih   = (const float*)d_in[6];
  const float* b_hh   = (const float*)d_in[7];
  const float* mlpW   = (const float*)d_in[8];
  const float* mlpb   = (const float*)d_in[9];
  const float* simW   = (const float*)d_in[10];
  const float* fcW    = (const float*)d_in[11];
  const float* fcb    = (const float*)d_in[12];

  char* ws = (char*)d_ws;
  size_t off = 0;
  auto alloc = [&](size_t bytes)->void*{
    void* p = ws + off; off += (bytes + 255) & ~(size_t)255; return p;
  };
  const size_t ROWS = (size_t)B_ * L_;                  // 32768
  bf16*     sae   = (bf16*)    alloc(ROWS * 512 * 2);
  bf16*     wih_b = (bf16*)    alloc((size_t)G4 * 512 * 2);
  bf16*     whh_b = (bf16*)    alloc((size_t)G4 * 512 * 2);
  bf16*     fcw_b = (bf16*)    alloc((size_t)1024 * 1024 * 2);
  float*    bsum  = (float*)   alloc((size_t)G4 * 4);
  float*    fcbp  = (float*)   alloc((size_t)1024 * 4);
  bf16*     xpre  = (bf16*)    alloc(ROWS * (size_t)G4 * 2);
  float*    outf  = (float*)   alloc(ROWS * (size_t)HID * 4);
  bf16*     hbuf  = (bf16*)    alloc((size_t)2 * 32 * 512 * 2);
  unsigned* cnt   = (unsigned*)alloc(256);
  float*    attb  = (float*)   alloc(ROWS * 4);
  bf16*     fin   = (bf16*)    alloc(ROWS * 1024 * 2);

  init_kernel<<<1, 256, 0, stream>>>(b_ih, b_hh, fcb, bsum, fcbp, hbuf, cnt);
  f32_to_bf16_k<<<1024, 256, 0, stream>>>(W_ih, wih_b, G4 * 512);
  f32_to_bf16_k<<<1024, 256, 0, stream>>>(W_hh, whh_b, G4 * 512);
  fcw_pad_k<<<1024, 256, 0, stream>>>(fcW, fcw_b);
  embed_kernel<<<32768, 256, 0, stream>>>(skill, answer, st, at, sae);

  // x_pre = sae @ W_ih^T + (b_ih+b_hh)   [M=32768, N=2048, K=512] -> bf16
  gemm_wmma<true, false><<<dim3(G4/128, 32768/128), 256, 40960, stream>>>(
      sae, wih_b, bsum, (void*)xpre, 32768, G4, 512, G4, G4);

  // persistent LSTM: 32 WGs, 160000 B dynamic LDS each
  lstm_kernel<<<32, 256, 160000, stream>>>(whh_b, xpre, hbuf, outf, cnt);

  att_kernel<<<32768/8, 256, 0, stream>>>(outf, mlpW, mlpb, simW, attb);
  scan_kernel<<<32, 256, 0, stream>>>(outf, attb, fin);

  // d_out = sigmoid(final @ fc_W^T + fc_b)  [M=32768, N=1024 padded, K=1024] -> f32, n<1000
  gemm_wmma<false, true><<<dim3(1024/128, 32768/128), 256, 40960, stream>>>(
      fin, fcw_b, fcbp, d_out, 32768, 1024, 1024, NUMC, NUMC);
}
